// MultiHeadAttention_85366769975869
// MI455X (gfx1250) — compile-verified, matrix operand order checked by reading, not run on hardware
//
#include <hip/hip_runtime.h>
#include <hip/hip_bf16.h>
#include <math.h>

// ---------------------------------------------------------------------------
// MI455X multi-head attention, bf16 WMMA path.
// ~70 GFLOP vs ~75MB HBM @ 23.3 TB/s -> strongly compute bound -> run every
// GEMM stage on v_wmma_f32_16x16x32_bf16 (f32 accumulate).
// ---------------------------------------------------------------------------

typedef __attribute__((ext_vector_type(16))) __bf16 v16bf;
typedef __attribute__((ext_vector_type(8)))  __bf16 v8bf;
typedef __attribute__((ext_vector_type(8)))  float  v8f;

#define BATCH   2
#define SEQ     2048
#define HEADS   16
#define DHEAD   64
#define DMODEL  1024
#define MTOT    (BATCH * SEQ)   // 4096

__device__ __forceinline__ v8f wmma_bf16(v16bf a, v16bf b, v8f c) {
    // D = A(16x32) * B(32x16) + C, f32 accum
    return __builtin_amdgcn_wmma_f32_16x16x32_bf16(
        /*neg_a=*/false, a, /*neg_b=*/false, b,
        /*c_mod=*/(short)0, c, /*reuse_a=*/false, /*reuse_b=*/false);
}

// Load a 16-row x 32-K bf16 fragment (A layout; B layout is symmetric:
// "row" = output column; both source matrices are row-major over K).
// Per ISA 7.12.2: lanes 0-15 hold rows 0-15 with K chunks {0-7,16-23},
// lanes 16-31 hold rows 0-15 with K chunks {8-15,24-31}.
__device__ __forceinline__ v16bf load_frag(const __bf16* p, int ld, int r0, int k0) {
    int lane = threadIdx.x & 31;
    const __bf16* base = p + (size_t)(r0 + (lane & 15)) * ld + k0 + ((lane >> 4) << 3);
    v8bf lo = *(const v8bf*)(base);
    v8bf hi = *(const v8bf*)(base + 16);
    v16bf f;
#pragma unroll
    for (int i = 0; i < 8; ++i) { f[i] = lo[i]; f[i + 8] = hi[i]; }
    return f;
}

// ---------------------------------------------------------------------------
// fp32 -> bf16 conversion
// ---------------------------------------------------------------------------
__global__ void cvt_bf16_kernel(const float* __restrict__ in,
                                __bf16* __restrict__ out, int n) {
    int i = blockIdx.x * blockDim.x + threadIdx.x;
    int stride = gridDim.x * blockDim.x;
    for (; i < n; i += stride) out[i] = (__bf16)in[i];
}

// ---------------------------------------------------------------------------
// C[M x N] = A[M x K] * B[N x K]^T   (both bf16 row-major over K)
// Block tile 128x128, K step 32. 8 waves as 4(M) x 2(N): each wave owns a
// 32x64 tile = 2x4 WMMA tiles -> 8 WMMA per 6 fragment loads per K step.
// MODE 0: write f32 row-major (ld = Nc) to Cout
// MODE 1: write bf16 scattered to per-head layout (b, h, n, d)
// ---------------------------------------------------------------------------
template <int MODE>
__global__ __launch_bounds__(256) void gemm_bf16_kernel(
    const __bf16* __restrict__ A, const __bf16* __restrict__ Bm,
    void* __restrict__ Cout, int M, int Nc, int K) {
    __shared__ __bf16 As[128 * 32];
    __shared__ __bf16 Bs[128 * 32];

    const int tid  = threadIdx.x;
    const int wave = tid >> 5;
    const int lane = tid & 31;
    const int wm   = wave & 3;      // 4 waves along M (32 rows each)
    const int wn   = wave >> 2;     // 2 waves along N (64 cols each)
    const int m0   = blockIdx.x * 128;
    const int n0   = blockIdx.y * 128;

    v8f acc[2][4] = {};

    for (int kt = 0; kt < K; kt += 32) {
        // Stage A and B slices: each 128x32 bf16, 16 elems/thread (2x b128)
        {
            int idx = tid * 16;
            int r = idx >> 5, c = idx & 31;
            const v8bf* srcA = (const v8bf*)(A + (size_t)(m0 + r) * K + kt + c);
            v8bf* dstA = (v8bf*)(As + idx);
            dstA[0] = srcA[0];
            dstA[1] = srcA[1];
            const v8bf* srcB = (const v8bf*)(Bm + (size_t)(n0 + r) * K + kt + c);
            v8bf* dstB = (v8bf*)(Bs + idx);
            dstB[0] = srcB[0];
            dstB[1] = srcB[1];
            if (kt + 32 < K) {
                __builtin_prefetch(A + (size_t)(m0 + r) * K + kt + 32 + c, 0, 0);
                __builtin_prefetch(Bm + (size_t)(n0 + r) * K + kt + 32 + c, 0, 0);
            }
        }
        __syncthreads();

        v16bf a0 = load_frag(As, 32, wm * 32, 0);
        v16bf a1 = load_frag(As, 32, wm * 32 + 16, 0);
#pragma unroll
        for (int t = 0; t < 4; ++t) {
            v16bf bt = load_frag(Bs, 32, wn * 64 + t * 16, 0);
            acc[0][t] = wmma_bf16(a0, bt, acc[0][t]);
            acc[1][t] = wmma_bf16(a1, bt, acc[1][t]);
        }
        __syncthreads();
    }

    // Epilogue. C/D layout: vgpr r, lane l -> (M = r + 8*(l>>4), N = l&15)
    const int cl = lane & 15, hl = lane >> 4;
#pragma unroll
    for (int i = 0; i < 2; ++i)
#pragma unroll
        for (int j = 0; j < 4; ++j)
#pragma unroll
            for (int r = 0; r < 8; ++r) {
                int mg = m0 + wm * 32 + i * 16 + r + 8 * hl;
                int ng = n0 + wn * 64 + j * 16 + cl;
                float v = acc[i][j][r];
                if (MODE == 0) {
                    ((float*)Cout)[(size_t)mg * Nc + ng] = v;
                } else {
                    int b = mg >> 11, n = mg & 2047;   // SEQ = 2048
                    int h = ng >> 6,  d = ng & 63;     // DHEAD = 64
                    size_t idx = ((((size_t)b * HEADS + h) * SEQ + n) << 6) + d;
                    ((__bf16*)Cout)[idx] = (__bf16)v;
                }
            }
}

// ---------------------------------------------------------------------------
// Causal flash attention per (b,h). Block = 256 threads = 8 waves,
// each wave owns 16 query rows; block covers 128 rows (so each staged V^T
// tile is reused by 8 waves). Key tiles of 32.
// Qh/Kh/Vh in (b,h,n,d) bf16. Output AO in (b*n, h*d) bf16 row-major.
// ---------------------------------------------------------------------------
__global__ __launch_bounds__(256) void flash_attn_kernel(
    const __bf16* __restrict__ Qh, const __bf16* __restrict__ Kh,
    const __bf16* __restrict__ Vh, __bf16* __restrict__ AO) {
    __shared__ __bf16 Vt[64 * 32];         // V^T tile: [d][key]
    __shared__ __bf16 Pbuf[8][16 * 32];    // per-wave P staging

    const int tid  = threadIdx.x;
    const int wave = tid >> 5;
    const int lane = tid & 31;
    const int cl = lane & 15, hl = lane >> 4;

    const int bh = blockIdx.y;              // b*HEADS + h
    const int b  = bh >> 4, h = bh & 15;
    const size_t head_off = (size_t)bh * SEQ * DHEAD;
    const __bf16* Q = Qh + head_off;
    const __bf16* K = Kh + head_off;
    const __bf16* V = Vh + head_off;

    const int q0    = blockIdx.x * 128;
    const int qrow0 = q0 + wave * 16;

    // Q fragments (16 rows x 64 = two K-chunks), reused across all key tiles
    v16bf aQ0 = load_frag(Q, DHEAD, qrow0, 0);
    v16bf aQ1 = load_frag(Q, DHEAD, qrow0, 32);

    float mi[8], li[8];
    v8f O[4] = {};
#pragma unroll
    for (int r = 0; r < 8; ++r) { mi[r] = -3.0e30f; li[r] = 0.0f; }

    const int jmax = (q0 + 127) >> 5;       // causal tile limit for the block
    for (int j = 0; j <= jmax; ++j) {
        __syncthreads();                    // protect Vt reuse
        // Stage V^T tile (32 keys x 64 d) -> Vt[d][key], 8 elems/thread
#pragma unroll
        for (int e = 0; e < 8; ++e) {
            int idx = tid + 256 * e;        // 0..2047
            int key = idx >> 6, d = idx & 63;
            Vt[d * 32 + key] = V[(size_t)(j * 32 + key) * DHEAD + d];
        }
        __syncthreads();

        // S = Q * K^T for this 16x32 slab (per wave)
        v8f s0 = {}, s1 = {};
        v16bf bK;
        bK = load_frag(K, DHEAD, j * 32,       0);  s0 = wmma_bf16(aQ0, bK, s0);
        bK = load_frag(K, DHEAD, j * 32,      32);  s0 = wmma_bf16(aQ1, bK, s0);
        bK = load_frag(K, DHEAD, j * 32 + 16,  0);  s1 = wmma_bf16(aQ0, bK, s1);
        bK = load_frag(K, DHEAD, j * 32 + 16, 32);  s1 = wmma_bf16(aQ1, bK, s1);

        // Wave-uniform: tile entirely below the diagonal for this wave's rows?
        const bool full = (j * 32 + 31) <= qrow0;

        // Online softmax; each lane owns rows (r + 8*hl), cols cl / cl+16
#pragma unroll
        for (int r = 0; r < 8; ++r) {
            float v0 = s0[r] * 0.125f;      // 1/sqrt(64)
            float v1 = s1[r] * 0.125f;
            if (!full) {                    // SALU branch: diagonal tiles only
                int row = qrow0 + r + 8 * hl;
                int c0 = j * 32 + cl;
                if (c0 > row)      v0 = -3.0e30f;
                if (c0 + 16 > row) v1 = -3.0e30f;
            }
            float mx = fmaxf(v0, v1);
#pragma unroll
            for (int off = 1; off < 16; off <<= 1)
                mx = fmaxf(mx, __shfl_xor(mx, off, 32));
            float mnew = fmaxf(mi[r], mx);
            float p0 = exp2f((v0 - mnew) * 1.44269504f);
            float p1 = exp2f((v1 - mnew) * 1.44269504f);
            float corr = exp2f((mi[r] - mnew) * 1.44269504f);
            float ps = p0 + p1;
#pragma unroll
            for (int off = 1; off < 16; off <<= 1)
                ps += __shfl_xor(ps, off, 32);
            li[r] = li[r] * corr + ps;
            mi[r] = mnew;
#pragma unroll
            for (int t = 0; t < 4; ++t) O[t][r] *= corr;
            int prow = r + 8 * hl;
            Pbuf[wave][prow * 32 + cl]      = (__bf16)p0;
            Pbuf[wave][prow * 32 + cl + 16] = (__bf16)p1;
        }

        // O += P(16x32) * V(32x64)   (wave-local LDS, DS ops are in-order)
        v16bf aP = load_frag(Pbuf[wave], 32, 0, 0);
#pragma unroll
        for (int t = 0; t < 4; ++t) {
            v16bf bV = load_frag(Vt, 32, t * 16, 0);
            O[t] = wmma_bf16(aP, bV, O[t]);
        }
    }

    // Normalize and write to (b*n, h*64+d) bf16
#pragma unroll
    for (int t = 0; t < 4; ++t)
#pragma unroll
        for (int r = 0; r < 8; ++r) {
            int row = qrow0 + r + 8 * hl;
            int col = h * DHEAD + t * 16 + cl;
            AO[(size_t)(b * SEQ + row) * DMODEL + col] = (__bf16)(O[t][r] / li[r]);
        }
}

// ---------------------------------------------------------------------------
// Host launch
// ---------------------------------------------------------------------------
extern "C" void kernel_launch(void* const* d_in, const int* in_sizes, int n_in,
                              void* d_out, int out_size, void* d_ws, size_t ws_size,
                              hipStream_t stream) {
    const float* q   = (const float*)d_in[0];
    const float* k   = (const float*)d_in[1];
    const float* v   = (const float*)d_in[2];
    const float* w_q = (const float*)d_in[3];
    const float* w_k = (const float*)d_in[4];
    const float* w_v = (const float*)d_in[5];
    const float* w_o = (const float*)d_in[6];

    char* ws = (char*)d_ws;
    size_t off = 0;
    auto alloc = [&](size_t bytes) -> void* {
        void* p = ws + off;
        off = (off + bytes + 255) & ~(size_t)255;
        return p;
    };

    const size_t NX = (size_t)MTOT * DMODEL;   // 4096*1024 activations
    const size_t NW = (size_t)DMODEL * DMODEL; // 1024*1024 weights

    __bf16* xq = (__bf16*)alloc(NX * 2);
    __bf16* xk = (__bf16*)alloc(NX * 2);
    __bf16* xv = (__bf16*)alloc(NX * 2);
    __bf16* wq = (__bf16*)alloc(NW * 2);
    __bf16* wk = (__bf16*)alloc(NW * 2);
    __bf16* wv = (__bf16*)alloc(NW * 2);
    __bf16* wo = (__bf16*)alloc(NW * 2);
    __bf16* Qh = (__bf16*)alloc(NX * 2);       // (b,h,n,d)
    __bf16* Kh = (__bf16*)alloc(NX * 2);
    __bf16* Vh = (__bf16*)alloc(NX * 2);
    __bf16* AO = (__bf16*)alloc(NX * 2);       // (b*n, h*d)

    // 1) fp32 -> bf16 conversions
    cvt_bf16_kernel<<<2048, 256, 0, stream>>>(q, xq, (int)NX);
    cvt_bf16_kernel<<<2048, 256, 0, stream>>>(k, xk, (int)NX);
    cvt_bf16_kernel<<<2048, 256, 0, stream>>>(v, xv, (int)NX);
    cvt_bf16_kernel<<<1024, 256, 0, stream>>>(w_q, wq, (int)NW);
    cvt_bf16_kernel<<<1024, 256, 0, stream>>>(w_k, wk, (int)NW);
    cvt_bf16_kernel<<<1024, 256, 0, stream>>>(w_v, wv, (int)NW);
    cvt_bf16_kernel<<<1024, 256, 0, stream>>>(w_o, wo, (int)NW);

    // 2) Q/K/V projections: (4096x1024) @ (1024x1024)^T -> head layout bf16
    dim3 ggrid(MTOT / 128, DMODEL / 128);
    gemm_bf16_kernel<1><<<ggrid, 256, 0, stream>>>(xq, wq, Qh, MTOT, DMODEL, DMODEL);
    gemm_bf16_kernel<1><<<ggrid, 256, 0, stream>>>(xk, wk, Kh, MTOT, DMODEL, DMODEL);
    gemm_bf16_kernel<1><<<ggrid, 256, 0, stream>>>(xv, wv, Vh, MTOT, DMODEL, DMODEL);

    // 3) Causal flash attention -> AO bf16 (b*n, h*d)
    flash_attn_kernel<<<dim3(SEQ / 128, BATCH * HEADS), 256, 0, stream>>>(Qh, Kh, Vh, AO);

    // 4) Output projection: AO @ w_o^T -> f32 d_out
    gemm_bf16_kernel<0><<<ggrid, 256, 0, stream>>>(AO, wo, d_out, MTOT, DMODEL, DMODEL);
}
